// groupvit_27367531610527
// MI455X (gfx1250) — matrix operand order repaired and checked by reading, not compiled, not run on hardware
//
#include <hip/hip_runtime.h>
#include <hip/hip_bf16.h>

#define NPTS 262144
#define SLOTS 64
#define DIM 256

typedef __attribute__((ext_vector_type(16))) __bf16 bf16x16;
typedef __attribute__((ext_vector_type(8)))  __bf16 bf16x8;
typedef __attribute__((ext_vector_type(8)))  float  f32x8;

__device__ __forceinline__ __bf16 f2bf(float f) { return (__bf16)f; }

__device__ __forceinline__ f32x8 zero8() {
  f32x8 z = {0.f,0.f,0.f,0.f,0.f,0.f,0.f,0.f};
  return z;
}

__device__ __forceinline__ f32x8 wmma_bf16(bf16x16 a, bf16x16 b, f32x8 c) {
  return __builtin_amdgcn_wmma_f32_16x16x32_bf16(false, a, false, b, (short)0, c, false, false);
}

// A fragment: 16x32 bf16 tile, row-major [m][k], stride lda (bf16 elems).
// Lane (m,lh) reads two contiguous 16B runs -> 2x ds_load_b128.
__device__ __forceinline__ bf16x16 frag_a_bf(const __bf16* A, int lda, int lane) {
  int m = lane & 15, lh = lane >> 4;
  const __bf16* p = A + m * lda + lh * 8;
  bf16x8 lo = *(const bf16x8*)(p);
  bf16x8 hi = *(const bf16x8*)(p + 16);
  return __builtin_shufflevector(lo, hi, 0,1,2,3,4,5,6,7,8,9,10,11,12,13,14,15);
}

// Same but f32 source (converted on load): 4x ds_load_b128 + cvt.
__device__ __forceinline__ bf16x16 frag_a_f32(const float* A, int lda, int lane) {
  int m = lane & 15, lh = lane >> 4;
  const float* p = A + m * lda + lh * 8;
  float4 a0 = ((const float4*)p)[0];
  float4 a1 = ((const float4*)p)[1];
  float4 b0 = ((const float4*)(p + 16))[0];
  float4 b1 = ((const float4*)(p + 16))[1];
  bf16x16 f;
  f[0]=f2bf(a0.x); f[1]=f2bf(a0.y); f[2]=f2bf(a0.z); f[3]=f2bf(a0.w);
  f[4]=f2bf(a1.x); f[5]=f2bf(a1.y); f[6]=f2bf(a1.z); f[7]=f2bf(a1.w);
  f[8]=f2bf(b0.x); f[9]=f2bf(b0.y); f[10]=f2bf(b0.z); f[11]=f2bf(b0.w);
  f[12]=f2bf(b1.x); f[13]=f2bf(b1.y); f[14]=f2bf(b1.z); f[15]=f2bf(b1.w);
  return f;
}

// B fragment: 32x16 tile B[k][n], but stored n-major S[n][k] with stride ldb.
// Lane (n,lh) reads two contiguous 16B runs -> 2x ds_load_b128.
__device__ __forceinline__ bf16x16 frag_b_nm(const __bf16* S, int ldb, int lane) {
  int n = lane & 15, lh = lane >> 4;
  const __bf16* p = S + n * ldb + lh * 8;
  bf16x8 lo = *(const bf16x8*)(p);
  bf16x8 hi = *(const bf16x8*)(p + 16);
  return __builtin_shufflevector(lo, hi, 0,1,2,3,4,5,6,7,8,9,10,11,12,13,14,15);
}

// ---------------- prep kernels ----------------

__global__ void prep_misc(const float* __restrict__ Wk, __bf16* __restrict__ wkbf,
                          float* __restrict__ mean_num, float* __restrict__ rowsum) {
  int i = blockIdx.x * 256 + threadIdx.x;   // 0..65535
  wkbf[i] = f2bf(Wk[i]);                    // plain bf16 copy, natural [e][d]
  if (i < SLOTS * DIM) mean_num[i] = 0.f;
  if (i < SLOTS) rowsum[i] = 0.f;
}

__global__ void prep_slots_q(const float* __restrict__ slots,
                             const float* __restrict__ g, const float* __restrict__ b,
                             const float* __restrict__ Wq, const float* __restrict__ bq,
                             __bf16* __restrict__ q_bf) {
  __shared__ float red[256];
  __shared__ float lns[256];
  int s = blockIdx.x, t = threadIdx.x;
  float x = slots[s * DIM + t];
  red[t] = x; __syncthreads();
  for (int o = 128; o > 0; o >>= 1) { if (t < o) red[t] += red[t + o]; __syncthreads(); }
  float mean = red[0] * (1.f / DIM);
  __syncthreads();
  float dv = x - mean;
  red[t] = dv * dv; __syncthreads();
  for (int o = 128; o > 0; o >>= 1) { if (t < o) red[t] += red[t + o]; __syncthreads(); }
  float var = red[0] * (1.f / DIM);
  float rs = rsqrtf(var + 1e-5f);
  lns[t] = dv * rs * g[t] + b[t];
  __syncthreads();
  float acc = bq[t];
  for (int d = 0; d < DIM; ++d) acc += lns[d] * Wq[t * DIM + d];
  q_bf[s * DIM + t] = f2bf(acc);            // q[s][e]
}

// ---------------- main fused pass ----------------
// grid 256 blocks, 256 threads, 8 sub-tiles of 128 points each.
__global__ __launch_bounds__(256)
void main_pass(const float* __restrict__ featg,
               const float* __restrict__ g, const float* __restrict__ b,
               const __bf16* __restrict__ wkbf, const float* __restrict__ bk,
               const __bf16* __restrict__ q_ws,
               float* __restrict__ attn1, float* __restrict__ attn2,
               float* __restrict__ mean_num, float* __restrict__ rowsum) {
  extern __shared__ char smem[];
  const int LDA = 272, RXT = 136, KNS = 272, QS = 272, AS = 132, WKS = 40;
  __bf16* lnx   = (__bf16*)smem;                 // [128][272] LN(x), A of GEMM1
  __bf16* rawxT = lnx   + 128 * 272;             // [256][136] raw x [d][n], B of GEMM3
  __bf16* knat  = rawxT + 256 * 136;             // [128][272] k [n][e], B of GEMM2
  __bf16* qlds  = knat  + 128 * 272;             // [64][272]  q, A of GEMM2
  float*  atl   = (float*)(qlds + 64 * 272);     // [64][132]  attn f32, A of GEMM3
  __bf16* wksn  = (__bf16*)(atl + 64 * 132);     // [256][40]  Wk strip [e][d0..d0+32], B of GEMM1

  int tid = threadIdx.x, wave = tid >> 5, lane = tid & 31;
  int nc = lane & 15, lh = lane >> 4;

  // q -> LDS once
  for (int i = tid; i < 64 * 128; i += 256) {
    int row = i >> 7, c = i & 127;
    ((unsigned int*)(qlds + row * QS))[c] = ((const unsigned int*)(q_ws + row * DIM))[c];
  }

  float gg[8], bb[8];
#pragma unroll
  for (int j = 0; j < 8; ++j) { gg[j] = g[lane * 8 + j]; bb[j] = b[lane * 8 + j]; }

  f32x8 acc3[8];                   // mean_num partial, carried across sub-tiles
#pragma unroll
  for (int t = 0; t < 8; ++t) acc3[t] = zero8();
  float rsum = 0.f;                // rowsum partial (tid<64)

  for (int sub = 0; sub < 8; ++sub) {
    long n0 = ((long)blockIdx.x * 8 + sub) * 128;
    __syncthreads();  // previous sub's GEMM3 done reading rawxT/atl

    // ---- stage 1: load rows + LayerNorm ----
    for (int rr = 0; rr < 16; ++rr) {
      int r = wave * 16 + rr;
      const float* src = featg + (n0 + r) * DIM + lane * 8;
      float v[8];
      float4 p0 = ((const float4*)src)[0];
      float4 p1 = ((const float4*)src)[1];
      v[0]=p0.x; v[1]=p0.y; v[2]=p0.z; v[3]=p0.w;
      v[4]=p1.x; v[5]=p1.y; v[6]=p1.z; v[7]=p1.w;
      float sum = 0.f, sq = 0.f;
#pragma unroll
      for (int j = 0; j < 8; ++j) { sum += v[j]; sq += v[j] * v[j]; }
#pragma unroll
      for (int o = 16; o > 0; o >>= 1) { sum += __shfl_xor(sum, o, 32); sq += __shfl_xor(sq, o, 32); }
      float mean = sum * (1.f / DIM);
      float var  = sq * (1.f / DIM) - mean * mean;
      float rs   = rsqrtf(var + 1e-5f);
      bf16x8 lv;
#pragma unroll
      for (int j = 0; j < 8; ++j) {
        float xv = v[j];
        rawxT[(lane * 8 + j) * RXT + r] = f2bf(xv);   // transposed [d][n]
        lv[j] = f2bf((xv - mean) * rs * gg[j] + bb[j]);
      }
      *(bf16x8*)(lnx + r * LDA + lane * 8) = lv;      // one ds_store_b128
    }

    // ---- GEMM1: k = lnx @ Wk^T (+bk) -> knat (bf16, [n][e]) ----
    {
      f32x8 acc[16];
#pragma unroll
      for (int j = 0; j < 16; ++j) acc[j] = zero8();
      for (int d0 = 0; d0 < DIM; d0 += 32) {
        __syncthreads();
        // stage Wk[e][d0..d0+32] strip (natural n-major layout for B)
        for (int i = tid; i < 256 * 16; i += 256) {
          int row = i >> 4, c = i & 15;
          ((unsigned int*)(wksn + row * WKS))[c] =
              ((const unsigned int*)(wkbf + row * DIM + d0))[c];
        }
        __syncthreads();
        bf16x16 af = frag_a_bf(lnx + (wave * 16) * LDA + d0, LDA, lane);
#pragma unroll
        for (int j = 0; j < 16; ++j) {
          bf16x16 bf = frag_b_nm(wksn + (j * 16) * WKS, WKS, lane);
          acc[j] = wmma_bf16(af, bf, acc[j]);
        }
      }
#pragma unroll
      for (int j = 0; j < 16; ++j) {
        int e = j * 16 + nc;
        float bkv = bk[e];
#pragma unroll
        for (int r = 0; r < 8; ++r) {
          int nl = wave * 16 + r + lh * 8;
          knat[nl * KNS + e] = f2bf(acc[j][r] + bkv);
        }
      }
    }
    __syncthreads();

    // ---- GEMM2: dots[s][n] = q . k ; softmax over s (registers) ----
    {
      f32x8 dacc[4];
#pragma unroll
      for (int mt = 0; mt < 4; ++mt) dacc[mt] = zero8();
      for (int e0 = 0; e0 < DIM; e0 += 32) {
        bf16x16 bf = frag_b_nm(knat + (wave * 16) * KNS + e0, KNS, lane);
#pragma unroll
        for (int mt = 0; mt < 4; ++mt) {
          bf16x16 af = frag_a_bf(qlds + (mt * 16) * QS + e0, QS, lane);
          dacc[mt] = wmma_bf16(af, bf, dacc[mt]);
        }
      }
      const float scale = 0.0625f;  // 256^-0.5
      float mx = -3.4e38f;
#pragma unroll
      for (int mt = 0; mt < 4; ++mt)
#pragma unroll
        for (int r = 0; r < 8; ++r) { float v = dacc[mt][r] * scale; dacc[mt][r] = v; mx = fmaxf(mx, v); }
      mx = fmaxf(mx, __shfl_xor(mx, 16, 32));
      float sm = 0.f;
#pragma unroll
      for (int mt = 0; mt < 4; ++mt)
#pragma unroll
        for (int r = 0; r < 8; ++r) { float e = __expf(dacc[mt][r] - mx); dacc[mt][r] = e; sm += e; }
      sm += __shfl_xor(sm, 16, 32);
      float inv = 1.f / sm;
#pragma unroll
      for (int mt = 0; mt < 4; ++mt)
#pragma unroll
        for (int r = 0; r < 8; ++r) {
          int s = mt * 16 + r + lh * 8;
          atl[s * AS + wave * 16 + nc] = dacc[mt][r] * inv;
        }
    }
    __syncthreads();

    // ---- write attn (both output copies), rowsum partial ----
    for (int i = tid; i < 64 * 128; i += 256) {
      int s = i >> 7, nl = i & 127;
      float v = atl[s * AS + nl];
      attn1[(long)s * NPTS + n0 + nl] = v;
      attn2[(long)s * NPTS + n0 + nl] = v;
    }
    if (tid < 64) {
      float ssum = 0.f;
      for (int nl = 0; nl < 128; ++nl) ssum += atl[tid * AS + nl];
      rsum += ssum;
    }

    // ---- GEMM3: mean_num partial += attn(bf16) @ rawx ----
    {
      int mt = wave & 3, dblk = (wave >> 2) * 8;
      for (int k0 = 0; k0 < 128; k0 += 32) {
        bf16x16 af = frag_a_f32(atl + (mt * 16) * AS + k0, AS, lane);
#pragma unroll
        for (int t = 0; t < 8; ++t) {
          bf16x16 bf = frag_b_nm(rawxT + ((dblk + t) * 16) * RXT + k0, RXT, lane);
          acc3[t] = wmma_bf16(af, bf, acc3[t]);
        }
      }
    }
  }

  // flush partials
  if (tid < 64) atomicAdd(&rowsum[tid], rsum);
  {
    int mt = wave & 3, dblk = (wave >> 2) * 8;
#pragma unroll
    for (int t = 0; t < 8; ++t) {
      int d = (dblk + t) * 16 + nc;
#pragma unroll
      for (int r = 0; r < 8; ++r) {
        int s = mt * 16 + r + lh * 8;
        atomicAdd(&mean_num[s * DIM + d], acc3[t][r]);
      }
    }
  }
}

// ---------------- finalize mean_slots ----------------
__global__ void finalize_ms(const float* __restrict__ mean_num, const float* __restrict__ rowsum,
                            float* __restrict__ out_ms, __bf16* __restrict__ ms_bfT) {
  int i = blockIdx.x * 256 + threadIdx.x;  // 16384, i = s*256 + d
  float v = mean_num[i] / (rowsum[i >> 8] + 1.f);
  out_ms[i] = v;
  ms_bfT[(i & 255) * SLOTS + (i >> 8)] = f2bf(v);   // transposed [d][s]
}

// ---------------- feat = attn^T @ mean_slots ----------------
__global__ __launch_bounds__(256)
void feat_pass(const float* __restrict__ attn1, const __bf16* __restrict__ ms_bfT,
               float* __restrict__ feat_out) {
  extern __shared__ char smem[];
  const int ATS = 72, MSS = 72;
  __bf16* atT   = (__bf16*)smem;             // [256][72]  attn^T [n][s], A operand
  __bf16* msl   = atT + 256 * 72;            // [256][72]  mean_slots^T [d][s], B operand (n-major)
  float*  stage = (float*)(msl + 256 * 72);  // 8 waves * [16][256] f32
  int tid = threadIdx.x, wave = tid >> 5, lane = tid & 31;
  int nc = lane & 15, lh = lane >> 4;
  long n0 = (long)blockIdx.x * 256;

  for (int i = tid; i < 64 * 256; i += 256) {
    int s = i >> 8, j = i & 255;
    atT[j * ATS + s] = f2bf(attn1[(long)s * NPTS + n0 + j]);
  }
  for (int i = tid; i < 256 * 32; i += 256) {
    int row = i >> 5, c = i & 31;
    ((unsigned int*)(msl + row * MSS))[c] = ((const unsigned int*)(ms_bfT + row * SLOTS))[c];
  }
  __syncthreads();

  float* sw = stage + wave * 16 * 256;
  for (int strip = wave; strip < 16; strip += 8) {
    int m0 = strip * 16;
    f32x8 acc[16];
#pragma unroll
    for (int j = 0; j < 16; ++j) acc[j] = zero8();
#pragma unroll
    for (int k0 = 0; k0 < 64; k0 += 32) {
      bf16x16 af = frag_a_bf(atT + m0 * ATS + k0, ATS, lane);
#pragma unroll
      for (int j = 0; j < 16; ++j) {
        bf16x16 bf = frag_b_nm(msl + (j * 16) * MSS + k0, MSS, lane);
        acc[j] = wmma_bf16(af, bf, acc[j]);
      }
    }
#pragma unroll
    for (int j = 0; j < 16; ++j)
#pragma unroll
      for (int r = 0; r < 8; ++r)
        sw[(r + lh * 8) * 256 + j * 16 + nc] = acc[j][r];
    // per-wave staging -> coalesced float4 stores
    for (int i = lane; i < 16 * 64; i += 32) {
      int r = i >> 6, c4 = i & 63;
      ((float4*)(feat_out + (n0 + m0 + r) * DIM))[c4] = ((const float4*)(sw + r * 256))[c4];
    }
  }
}

// ---------------- launch ----------------
extern "C" void kernel_launch(void* const* d_in, const int* in_sizes, int n_in,
                              void* d_out, int out_size, void* d_ws, size_t ws_size,
                              hipStream_t stream) {
  (void)in_sizes; (void)n_in; (void)out_size; (void)ws_size;
  const float* featg = (const float*)d_in[0];
  const float* slots = (const float*)d_in[1];
  const float* g_in  = (const float*)d_in[2];
  const float* b_in  = (const float*)d_in[3];
  const float* g_s   = (const float*)d_in[4];
  const float* b_s   = (const float*)d_in[5];
  const float* Wq    = (const float*)d_in[6];
  const float* bq    = (const float*)d_in[7];
  const float* Wk    = (const float*)d_in[8];
  const float* bk    = (const float*)d_in[9];

  float* out       = (float*)d_out;
  float* out_feat  = out;
  float* out_attn1 = out + (size_t)NPTS * DIM;
  float* out_attn2 = out_attn1 + (size_t)SLOTS * NPTS;
  float* out_ms    = out_attn2 + (size_t)SLOTS * NPTS;

  char* ws = (char*)d_ws;
  __bf16* wkbf     = (__bf16*)ws;              // 131072 B  (bf16 copy of Wk, [e][d])
  __bf16* q_bf     = (__bf16*)(ws + 131072);   // 32768 B
  __bf16* ms_bfT   = (__bf16*)(ws + 163840);   // 32768 B   ([d][s])
  float*  mean_num = (float*)(ws + 196608);    // 65536 B
  float*  rowsum   = (float*)(ws + 262144);    // 256 B

  prep_misc<<<256, 256, 0, stream>>>(Wk, wkbf, mean_num, rowsum);
  prep_slots_q<<<64, 256, 0, stream>>>(slots, g_s, b_s, Wq, bq, q_bf);

  // LDS: lnx 69632 + rawxT 69632 + knat 69632 + q 34816 + attn 33792 + wk strip 20480
  size_t lds_main = 297984;  // bytes, < 320KB WGP limit
  main_pass<<<NPTS / (128 * 8), 256, lds_main, stream>>>(
      featg, g_in, b_in, wkbf, bk, q_bf, out_attn1, out_attn2, mean_num, rowsum);

  finalize_ms<<<64, 256, 0, stream>>>(mean_num, rowsum, out_ms, ms_bfT);

  size_t lds_feat = 36864 + 36864 + 131072;  // 204800 bytes
  feat_pass<<<NPTS / 256, 256, lds_feat, stream>>>(out_attn1, ms_bfT, out_feat);
}
